// MambaGR_49941879718202
// MI455X (gfx1250) — compile-verified, matrix operand order checked by reading, not run on hardware
//
#include <hip/hip_runtime.h>
#include <hip/hip_bf16.h>
#include <math.h>

#define BATCH   16
#define TSEQ    2048
#define BT      (BATCH*TSEQ)      // 32768 tokens
#define DMODEL  288
#define NLAYERS 10
#define DSTATE  144
#define CONVSZ  4
#define DINNER  576
#define GRUH    256
#define DTRANK  18
#define DTK     32                // padded dt rank
#define XPN     320               // padded x_proj out (306 -> 320)
#define DIN     768
#define GRU3H   768               // 3*GRUH

typedef __bf16 bf16_t;
typedef __attribute__((ext_vector_type(8)))  bf16_t v8bf;
typedef __attribute__((ext_vector_type(16))) bf16_t v16bf;
typedef __attribute__((ext_vector_type(8)))  float  v8f;
typedef __attribute__((ext_vector_type(4)))  unsigned int u32x4;
typedef __attribute__((ext_vector_type(4)))  int i32x4;
typedef __attribute__((ext_vector_type(8)))  int i32x8;

#if defined(__has_builtin)
#if __has_builtin(__builtin_amdgcn_tensor_load_to_lds) && __has_builtin(__builtin_amdgcn_s_wait_tensorcnt)
#define HAVE_TDM 1
#endif
#endif

// ---------------------------------------------------------------------------
// Fragment gather per CDNA5 ISA 16-bit A/B layout (CDNA5 ISA):
// lanes 0-15:  K = [k..k+7] and [k+16..k+23]
// lanes 16-31: K = [k+8..k+15] and [k+24..k+31]
// Two 16B loads (global_load_b128 or ds_load_b128).
// ---------------------------------------------------------------------------
__device__ __forceinline__ v16bf frag_load(const bf16_t* rowk, int hx) {
    const bf16_t* base = rowk + hx * 8;
    v8bf lo = *(const v8bf*)(base);
    v8bf hi = *(const v8bf*)(base + 16);
    return __builtin_shufflevector(lo, hi, 0,1,2,3,4,5,6,7,8,9,10,11,12,13,14,15);
}

__device__ __forceinline__ float sigmf(float x) { return 1.f / (1.f + __expf(-x)); }

// ---------------------------------------------------------------------------
// Generic WMMA GEMM:  C[M,N] (f32) = act( A[M,K](bf16) @ W[N,K](bf16)^T + bias )
// One wave per 16x32 output tile pair. M mult of 16, N mult of 32, K mult of 32.
// ACT: 0=none, 1=softplus, 2=sigmoid.  ADDC: C += result.
// ---------------------------------------------------------------------------
template <int ACT, bool ADDC>
__global__ __launch_bounds__(32) void k_gemm_wmma(
    const bf16_t* __restrict__ A, const bf16_t* __restrict__ W,
    const float* __restrict__ bias, float* __restrict__ C, int K, int N)
{
    const int lane = threadIdx.x & 31;
    const int r  = lane & 15;
    const int hx = lane >> 4;
    const int tm  = blockIdx.y;
    const int tn0 = blockIdx.x * 2;

    const bf16_t* arow = A + (size_t)(tm * 16 + r) * K;
    const bf16_t* w0   = W + (size_t)(tn0 * 16 + r) * K;
    const bf16_t* w1   = w0 + (size_t)16 * K;

    v8f acc0 = {}; v8f acc1 = {};
    for (int k = 0; k < K; k += 32) {
        __builtin_prefetch(arow + k + 128, 0, 1);   // global_prefetch_b8
        __builtin_prefetch(w0   + k + 128, 0, 1);
        v16bf a  = frag_load(arow + k, hx);
        v16bf b0 = frag_load(w0   + k, hx);
        v16bf b1 = frag_load(w1   + k, hx);
        acc0 = __builtin_amdgcn_wmma_f32_16x16x32_bf16(false, a, false, b0,
                                                       (short)0, acc0, false, false);
        acc1 = __builtin_amdgcn_wmma_f32_16x16x32_bf16(false, a, false, b1,
                                                       (short)0, acc1, false, false);
    }

    const int n0 = tn0 * 16 + r;
    const int n1 = n0 + 16;
    const int m0 = tm * 16 + hx * 8;   // C/D layout: vgpr i -> M = m0+i
    const float bi0 = bias ? bias[n0] : 0.f;
    const float bi1 = bias ? bias[n1] : 0.f;

#pragma unroll
    for (int i = 0; i < 8; ++i) {
        float v0 = acc0[i] + bi0;
        float v1 = acc1[i] + bi1;
        if constexpr (ACT == 1) {                    // softplus
            v0 = (v0 > 20.f) ? v0 : log1pf(__expf(v0));
            v1 = (v1 > 20.f) ? v1 : log1pf(__expf(v1));
        } else if constexpr (ACT == 2) {             // sigmoid
            v0 = sigmf(v0);
            v1 = sigmf(v1);
        }
        size_t row = (size_t)(m0 + i) * N;
        if constexpr (ADDC) { C[row + n0] += v0; C[row + n1] += v1; }
        else                { C[row + n0]  = v0; C[row + n1]  = v1; }
    }
}

// ---------------------------------------------------------------------------
// Elementwise / conversion kernels
// ---------------------------------------------------------------------------
__global__ void k_f32_to_bf16(const float* __restrict__ s, bf16_t* __restrict__ d, size_t n) {
    size_t i  = blockIdx.x * (size_t)blockDim.x + threadIdx.x;
    size_t st = (size_t)gridDim.x * blockDim.x;
    for (; i < n; i += st) d[i] = (bf16_t)s[i];
}

__global__ void k_cvt_pad_rows(const float* __restrict__ s, bf16_t* __restrict__ d,
                               int srcRows, int K, int dstRows) {
    size_t n  = (size_t)dstRows * K;
    size_t st = (size_t)gridDim.x * blockDim.x;
    for (size_t i = blockIdx.x * (size_t)blockDim.x + threadIdx.x; i < n; i += st) {
        size_t row = i / K, c = i - row * K;
        d[i] = (row < (size_t)srcRows) ? (bf16_t)s[row * K + c] : (bf16_t)0.f;
    }
}

__global__ void k_cvt_pad_cols(const float* __restrict__ s, bf16_t* __restrict__ d,
                               int rows, int Ks, int Kd) {
    size_t n  = (size_t)rows * Kd;
    size_t st = (size_t)gridDim.x * blockDim.x;
    for (size_t i = blockIdx.x * (size_t)blockDim.x + threadIdx.x; i < n; i += st) {
        size_t row = i / Kd, c = i - row * Kd;
        d[i] = (c < (size_t)Ks) ? (bf16_t)s[row * Ks + c] : (bf16_t)0.f;
    }
}

__global__ void k_zero_f32(float* __restrict__ p, int n) {
    int i = blockIdx.x * blockDim.x + threadIdx.x;
    if (i < n) p[i] = 0.f;
}

// depthwise causal conv (window 4) + bias + SiLU; u read from xz[:, :DINNER]
__global__ void k_conv_silu(const float* __restrict__ xz, const float* __restrict__ cw,
                            const float* __restrict__ cb, float* __restrict__ uact,
                            bf16_t* __restrict__ ucbf) {
    size_t idx = blockIdx.x * (size_t)blockDim.x + threadIdx.x;
    if (idx >= (size_t)BT * DINNER) return;
    int    d  = (int)(idx % DINNER);
    size_t bt = idx / DINNER;
    int    t  = (int)(bt % TSEQ);
    size_t b  = bt / TSEQ;
    float s = cb[d];
#pragma unroll
    for (int j = 0; j < CONVSZ; ++j) {
        int tt = t - (CONVSZ - 1) + j;
        if (tt >= 0)
            s += cw[d * CONVSZ + j] * xz[((size_t)b * TSEQ + tt) * (2 * DINNER) + d];
    }
    float si = s * sigmf(s);
    uact[idx] = si;
    ucbf[idx] = (bf16_t)si;
}

// extract dt_r (cols 0..17 of x_dbl) into zero-padded bf16 (BT x 32)
__global__ void k_dtr_pad(const float* __restrict__ xdbl, bf16_t* __restrict__ dtrbf) {
    size_t idx = blockIdx.x * (size_t)blockDim.x + threadIdx.x;
    if (idx >= (size_t)BT * DTK) return;
    int    c  = (int)(idx % DTK);
    size_t bt = idx / DTK;
    dtrbf[idx] = (c < DTRANK) ? (bf16_t)xdbl[bt * XPN + c] : (bf16_t)0.f;
}

// Selective scan: one thread per (b,d); A and h state fully unrolled in regs.
// Fuses  y = (scan + u*Dskip) * silu(z)  and writes bf16 for out_proj.
__global__ void k_scan(const float* __restrict__ dtf,  const float* __restrict__ uact,
                       const float* __restrict__ xdbl, const float* __restrict__ xz,
                       const float* __restrict__ A_log, const float* __restrict__ Dskip,
                       bf16_t* __restrict__ ybf) {
    int idx = blockIdx.x * blockDim.x + threadIdx.x;
    if (idx >= BATCH * DINNER) return;
    int b = idx / DINNER, d = idx - b * DINNER;

    float Areg[DSTATE];
#pragma unroll
    for (int n = 0; n < DSTATE; ++n) Areg[n] = -__expf(A_log[(size_t)d * DSTATE + n]);
    float hst[DSTATE];
#pragma unroll
    for (int n = 0; n < DSTATE; ++n) hst[n] = 0.f;
    const float Dsk = Dskip[d];

    for (int t = 0; t < TSEQ; ++t) {
        size_t bt  = (size_t)b * TSEQ + t;
        float dtv  = dtf[bt * DINNER + d];
        float uv   = uact[bt * DINNER + d];
        float zv   = xz[bt * (2 * DINNER) + DINNER + d];
        const float* Brow = xdbl + bt * XPN + DTRANK;     // lane-uniform (broadcast)
        const float* Crow = Brow + DSTATE;
        float du = dtv * uv;
        float y  = 0.f;
#pragma unroll
        for (int n = 0; n < DSTATE; ++n) {
            float hv = __expf(dtv * Areg[n]) * hst[n] + du * Brow[n];
            hst[n] = hv;
            y += hv * Crow[n];
        }
        y += uv * Dsk;
        ybf[bt * DINNER + d] = (bf16_t)(y * (zv * sigmf(zv)));
    }
}

// h = h * g, and refresh bf16 copy of h for the next layer's WMMA GEMMs
__global__ void k_mul_gate(float* __restrict__ h, const float* __restrict__ g,
                           bf16_t* __restrict__ hbf, size_t n) {
    size_t i  = blockIdx.x * (size_t)blockDim.x + threadIdx.x;
    size_t st = (size_t)gridDim.x * blockDim.x;
    for (; i < n; i += st) {
        float v = h[i] * g[i];
        h[i]   = v;
        hbf[i] = (bf16_t)v;
    }
}

// row L2-normalize (wave per 768-wide row), wave32 shuffle reduction
__global__ void k_normalize(float* __restrict__ out) {
    int row  = blockIdx.x * 8 + (threadIdx.x >> 5);
    int lane = threadIdx.x & 31;
    float* p = out + (size_t)row * DIN;
    float s = 0.f;
    for (int c = lane; c < DIN; c += 32) { float v = p[c]; s += v * v; }
#pragma unroll
    for (int m = 16; m; m >>= 1) s += __shfl_xor(s, m, 32);
    float inv = 1.f / fmaxf(sqrtf(s), 1e-12f);
    for (int c = lane; c < DIN; c += 32) p[c] *= inv;
}

// ---------------------------------------------------------------------------
// Persistent sequential GRU (gi precomputed by parallel WMMA GEMM).
// 512 threads = 16 waves = 4 waves/SIMD, which unlocks the full 256-VGPR/wave
// budget (24 waves capped us at ~160 and forced a weight-fragment spill).
// Each wave owns 3 of the 48 gh N-tiles: 24 w_hh fragments = 192 VGPRs,
// constant-indexed, register-resident; each ds_load_b128 A-fragment pair
// feeds 3 back-to-back v_wmma.
// Wave 0 DMAs the 16x768 f32 gi tile for step t into LDS with the Tensor
// Data Mover, overlapping the WMMA gh computation; s_wait_tensorcnt + barrier
// publish it before the gate math.
// ---------------------------------------------------------------------------
#define GH_OFF    0
#define HF_OFF    (16*GRU3H*4)                 // 49152
#define HB_OFF    (HF_OFF + 16*GRUH*4)         // 65536
#define GIBUF_OFF (HB_OFF + 16*GRUH*2)         // 73728
#define GRU_SMEM  (GIBUF_OFF + 16*GRU3H*4)     // 122880 B
#define GRU_THREADS 512

__global__ __launch_bounds__(GRU_THREADS, 1) void k_gru_seq(
    const float* __restrict__ giL,   // (16,2048,768) precomputed gi (+bias)
    const bf16_t* __restrict__ whh,  // (768,256) bf16
    const float* __restrict__ bhh,
    float* __restrict__ hprev_g,     // (16,256) persistent across layers
    bf16_t* __restrict__ hsbf)       // (16,2048,256) hidden states, bf16
{
    extern __shared__ char smem[];
    float*  Gh    = (float*)(smem + GH_OFF);     // 16 x 768
    float*  Hf    = (float*)(smem + HF_OFF);     // 16 x 256 (f32 state)
    bf16_t* Hb    = (bf16_t*)(smem + HB_OFF);    // 16 x 256 (bf16 state)
    float*  GiBuf = (float*)(smem + GIBUF_OFF);  // 16 x 768 staged gi tile

    const int tid  = threadIdx.x;
    const int lane = tid & 31;
    const int wave = tid >> 5;                   // 0..15
    const int r    = lane & 15;
    const int hx   = lane >> 4;

    for (int i = tid; i < 16 * GRUH; i += GRU_THREADS) {
        float v = hprev_g[i];
        Hf[i] = v;
        Hb[i] = (bf16_t)v;
    }
    __syncthreads();

    // preload this wave's three w_hh tiles into registers (constant indices)
    const int n0 = (wave * 3 + 0) * 16 + r;
    const int n1 = (wave * 3 + 1) * 16 + r;
    const int n2 = (wave * 3 + 2) * 16 + r;
    v16bf wf0[8], wf1[8], wf2[8];
    {
        const bf16_t* w0 = whh + (size_t)n0 * GRUH;
        const bf16_t* w1 = whh + (size_t)n1 * GRUH;
        const bf16_t* w2 = whh + (size_t)n2 * GRUH;
#pragma unroll
        for (int k = 0; k < 8; ++k) {
            wf0[k] = frag_load(w0 + k * 32, hx);
            wf1[k] = frag_load(w1 + k * 32, hx);
            wf2[k] = frag_load(w2 + k * 32, hx);
        }
    }
    const float bh0 = bhh[n0];
    const float bh1 = bhh[n1];
    const float bh2 = bhh[n2];

    for (int t = 0; t < TSEQ; ++t) {
#ifdef HAVE_TDM
        if (wave == 0) {
            // TDM: DMA 2D tile (tile_dim0=768 contiguous f32, 16 rows with
            // row stride TSEQ*768 elements) from gi[:, t, :] into GiBuf.
            unsigned long long ga =
                (unsigned long long)(uintptr_t)(giL + (size_t)t * GRU3H);
            u32x4 g0;
            g0.x = 1u;                                   // count=1, user D#
            g0.y = (unsigned)GIBUF_OFF;                  // lds_addr (bytes)
            g0.z = (unsigned)(ga & 0xFFFFFFFFu);         // global_addr[31:0]
            g0.w = (unsigned)((ga >> 32) & 0x1FFFFFFu)   // global_addr[56:32]
                   | 0x80000000u;                        // type=2 ("image")
            i32x8 g1;
            g1[0] = (int)(2u << 16);          // wg_mask=0, data_size=4B
            g1[1] = (int)((unsigned)GRU3H << 16);   // tensor_dim0[15:0] @ bits[63:48]
            g1[2] = (int)(16u << 16);         // tensor_dim0 hi=0, tensor_dim1=16
            g1[3] = (int)((unsigned)GRU3H << 16);   // tensor_dim1 hi=0, tile_dim0=768
            g1[4] = 16;                       // tile_dim1=16, tile_dim2=0
            g1[5] = (int)(TSEQ * GRU3H);      // tensor_dim0_stride[31:0]
            g1[6] = 0;
            g1[7] = 0;
            i32x4 z4 = {0, 0, 0, 0};
#if __clang_major__ >= 23
            i32x8 z8 = {0, 0, 0, 0, 0, 0, 0, 0};
            __builtin_amdgcn_tensor_load_to_lds(g0, g1, z4, z4, z8, 0);
#else
            __builtin_amdgcn_tensor_load_to_lds(g0, g1, z4, z4, 0);
#endif
        }
#endif
        // recurrent matmul: 3 gh tiles for this wave (overlaps the TDM DMA)
        v8f a0 = {}; v8f a1 = {}; v8f a2 = {};
#pragma unroll
        for (int k = 0; k < 8; ++k) {
            v16bf av = frag_load(Hb + r * GRUH + k * 32, hx);   // ds_load_b128
            a0 = __builtin_amdgcn_wmma_f32_16x16x32_bf16(false, av, false, wf0[k],
                                                         (short)0, a0, false, false);
            a1 = __builtin_amdgcn_wmma_f32_16x16x32_bf16(false, av, false, wf1[k],
                                                         (short)0, a1, false, false);
            a2 = __builtin_amdgcn_wmma_f32_16x16x32_bf16(false, av, false, wf2[k],
                                                         (short)0, a2, false, false);
        }
#pragma unroll
        for (int i = 0; i < 8; ++i) {
            Gh[(hx * 8 + i) * GRU3H + n0] = a0[i] + bh0;
            Gh[(hx * 8 + i) * GRU3H + n1] = a1[i] + bh1;
            Gh[(hx * 8 + i) * GRU3H + n2] = a2[i] + bh2;
        }
#ifdef HAVE_TDM
        if (wave == 0) __builtin_amdgcn_s_wait_tensorcnt(0);
#endif
        __syncthreads();

        // gates + state update (4096 elems over 512 threads)
        for (int e = tid; e < 16 * GRUH; e += GRU_THREADS) {
            int b = e >> 8, j = e & 255;
#ifdef HAVE_TDM
            float ir  = GiBuf[b * GRU3H + j];
            float iz  = GiBuf[b * GRU3H + 256 + j];
            float inn = GiBuf[b * GRU3H + 512 + j];
#else
            size_t gbase = ((size_t)b * TSEQ + t) * GRU3H;
            float ir  = giL[gbase + j];
            float iz  = giL[gbase + 256 + j];
            float inn = giL[gbase + 512 + j];
#endif
            float hr  = Gh[b * GRU3H + j];
            float hz  = Gh[b * GRU3H + 256 + j];
            float hn  = Gh[b * GRU3H + 512 + j];
            float rr = sigmf(ir + hr);
            float zz = sigmf(iz + hz);
            float nn = tanhf(inn + rr * hn);
            float hnew = (1.f - zz) * nn + zz * Hf[b * GRUH + j];
            Hf[b * GRUH + j] = hnew;
            hsbf[((size_t)b * TSEQ + t) * GRUH + j] = (bf16_t)hnew;
        }
        __syncthreads();
        for (int e = tid; e < 16 * GRUH; e += GRU_THREADS) Hb[e] = (bf16_t)Hf[e];
        __syncthreads();
    }

    for (int i = tid; i < 16 * GRUH; i += GRU_THREADS) hprev_g[i] = Hf[i];
}

// ---------------------------------------------------------------------------
// Host orchestration
// ---------------------------------------------------------------------------
extern "C" void kernel_launch(void* const* d_in, const int* in_sizes, int n_in,
                              void* d_out, int out_size, void* d_ws, size_t ws_size,
                              hipStream_t stream) {
    (void)in_sizes; (void)n_in; (void)out_size; (void)ws_size;

    const float* x     = (const float*)d_in[0];
    const float* ipw   = (const float*)d_in[1];
    const float* ipb   = (const float*)d_in[2];
    const float* headw = (const float*)d_in[3];
    const float* inw   = (const float*)d_in[4];
    const float* inb   = (const float*)d_in[5];
    const float* convw = (const float*)d_in[6];
    const float* convb = (const float*)d_in[7];
    const float* xpw   = (const float*)d_in[8];
    const float* dtw   = (const float*)d_in[9];
    const float* dtb   = (const float*)d_in[10];
    const float* alog  = (const float*)d_in[11];
    const float* dskip = (const float*)d_in[12];
    const float* opw   = (const float*)d_in[13];
    const float* opb   = (const float*)d_in[14];
    const float* wih   = (const float*)d_in[15];
    const float* whh   = (const float*)d_in[16];
    const float* bih   = (const float*)d_in[17];
    const float* bhh   = (const float*)d_in[18];
    const float* gw    = (const float*)d_in[19];
    float* out = (float*)d_out;

    // workspace carve-out (256B aligned segments)
    char* p = (char*)d_ws;
    auto take = [&](size_t bytes) -> char* {
        char* r = p;
        p += (bytes + 255) & ~(size_t)255;
        return r;
    };
    bf16_t* xbf    = (bf16_t*)take((size_t)BT * DIN * 2);
    float*  h      = (float*) take((size_t)BT * DMODEL * 4);
    bf16_t* hbf    = (bf16_t*)take((size_t)BT * DMODEL * 2);
    float*  xz     = (float*) take((size_t)BT * 2 * DINNER * 4);
    float*  uact   = (float*) take((size_t)BT * DINNER * 4);
    bf16_t* ucbf   = (bf16_t*)take((size_t)BT * DINNER * 2);
    float*  xdbl   = (float*) take((size_t)BT * XPN * 4);
    bf16_t* dtrbf  = (bf16_t*)take((size_t)BT * DTK * 2);
    float*  dtf    = (float*) take((size_t)BT * DINNER * 4);
    bf16_t* ybf    = (bf16_t*)take((size_t)BT * DINNER * 2);
    float*  gi_all = (float*) take((size_t)BT * GRU3H * 4);
    bf16_t* hsbf   = (bf16_t*)take((size_t)BT * GRUH * 2);
    float*  gbuf   = (float*) take((size_t)BT * DMODEL * 4);
    float*  hprev  = (float*) take((size_t)BATCH * GRUH * 4);
    // bf16 weight copies (L2-resident)
    bf16_t* ipw_b   = (bf16_t*)take((size_t)DMODEL * DIN * 2);
    bf16_t* headw_b = (bf16_t*)take((size_t)DIN * DMODEL * 2);
    bf16_t* inw_b   = (bf16_t*)take((size_t)NLAYERS * 2 * DINNER * DMODEL * 2);
    bf16_t* xpw_b   = (bf16_t*)take((size_t)NLAYERS * XPN * DINNER * 2);
    bf16_t* dtw_b   = (bf16_t*)take((size_t)NLAYERS * DINNER * DTK * 2);
    bf16_t* opw_b   = (bf16_t*)take((size_t)NLAYERS * DMODEL * DINNER * 2);
    bf16_t* wih_b   = (bf16_t*)take((size_t)NLAYERS * GRU3H * DMODEL * 2);
    bf16_t* whh_b   = (bf16_t*)take((size_t)NLAYERS * GRU3H * GRUH * 2);
    bf16_t* gw_b    = (bf16_t*)take((size_t)NLAYERS * DMODEL * GRUH * 2);

    auto CV = [&](const float* s, bf16_t* d, size_t n) {
        k_f32_to_bf16<<<4096, 256, 0, stream>>>(s, d, n);
    };

    // one-time conversions (per launch; deterministic)
    CV(x,     xbf,     (size_t)BT * DIN);
    CV(ipw,   ipw_b,   (size_t)DMODEL * DIN);
    CV(headw, headw_b, (size_t)DIN * DMODEL);
    CV(inw,   inw_b,   (size_t)NLAYERS * 2 * DINNER * DMODEL);
    CV(opw,   opw_b,   (size_t)NLAYERS * DMODEL * DINNER);
    CV(wih,   wih_b,   (size_t)NLAYERS * GRU3H * DMODEL);
    CV(whh,   whh_b,   (size_t)NLAYERS * GRU3H * GRUH);
    CV(gw,    gw_b,    (size_t)NLAYERS * DMODEL * GRUH);
    for (int L = 0; L < NLAYERS; ++L) {
        k_cvt_pad_rows<<<1024, 256, 0, stream>>>(
            xpw + (size_t)L * (DTRANK + 2 * DSTATE) * DINNER,
            xpw_b + (size_t)L * XPN * DINNER, DTRANK + 2 * DSTATE, DINNER, XPN);
        k_cvt_pad_cols<<<128, 256, 0, stream>>>(
            dtw + (size_t)L * DINNER * DTRANK,
            dtw_b + (size_t)L * DINNER * DTK, DINNER, DTRANK, DTK);
    }
    k_zero_f32<<<(BATCH * GRUH + 255) / 256, 256, 0, stream>>>(hprev, BATCH * GRUH);

    // input projection: h = x @ ipw^T + ipb
    k_gemm_wmma<0, false><<<dim3(DMODEL / 32, BT / 16), 32, 0, stream>>>(
        xbf, ipw_b, ipb, h, DIN, DMODEL);
    CV(h, hbf, (size_t)BT * DMODEL);

    for (int L = 0; L < NLAYERS; ++L) {
        // 1) in_proj: xz = h @ in_w^T + in_b   (BT x 1152)
        k_gemm_wmma<0, false><<<dim3((2 * DINNER) / 32, BT / 16), 32, 0, stream>>>(
            hbf, inw_b + (size_t)L * 2 * DINNER * DMODEL,
            inb + (size_t)L * 2 * DINNER, xz, DMODEL, 2 * DINNER);
        // 2) depthwise causal conv + SiLU
        k_conv_silu<<<((size_t)BT * DINNER + 255) / 256, 256, 0, stream>>>(
            xz, convw + (size_t)L * DINNER * CONVSZ, convb + (size_t)L * DINNER,
            uact, ucbf);
        // 3) x_proj: x_dbl = u @ xp_w^T   (BT x 320, zero-padded cols)
        k_gemm_wmma<0, false><<<dim3(XPN / 32, BT / 16), 32, 0, stream>>>(
            ucbf, xpw_b + (size_t)L * XPN * DINNER, nullptr, xdbl, DINNER, XPN);
        // 4) dt_r extraction -> zero-padded bf16 (BT x 32)
        k_dtr_pad<<<((size_t)BT * DTK + 255) / 256, 256, 0, stream>>>(xdbl, dtrbf);
        // 5) dt = softplus(dt_r @ dt_w^T + dt_b)
        k_gemm_wmma<1, false><<<dim3(DINNER / 32, BT / 16), 32, 0, stream>>>(
            dtrbf, dtw_b + (size_t)L * DINNER * DTK, dtb + (size_t)L * DINNER,
            dtf, DTK, DINNER);
        // 6) selective scan (fused + u*Dskip and *silu(z)), writes bf16 y
        k_scan<<<(BATCH * DINNER + 255) / 256, 256, 0, stream>>>(
            dtf, uact, xdbl, xz, alog + (size_t)L * DINNER * DSTATE,
            dskip + (size_t)L * DINNER, ybf);
        // 7) out_proj (residual): h += y @ op_w^T + op_b
        k_gemm_wmma<0, true><<<dim3(DMODEL / 32, BT / 16), 32, 0, stream>>>(
            ybf, opw_b + (size_t)L * DMODEL * DINNER, opb + (size_t)L * DMODEL,
            h, DINNER, DMODEL);
        // 8) refresh bf16 h
        CV(h, hbf, (size_t)BT * DMODEL);
        // 9) gi = h @ w_ih^T + b_ih for ALL timesteps (parallel WMMA GEMM)
        k_gemm_wmma<0, false><<<dim3(GRU3H / 32, BT / 16), 32, 0, stream>>>(
            hbf, wih_b + (size_t)L * GRU3H * DMODEL, bih + (size_t)L * GRU3H,
            gi_all, DMODEL, GRU3H);
        // 10) sequential GRU recurrence (gh only) with TDM-staged gi tiles
        k_gru_seq<<<1, GRU_THREADS, GRU_SMEM, stream>>>(
            gi_all, whh_b + (size_t)L * GRU3H * GRUH, bhh + (size_t)L * GRU3H,
            hprev, hsbf);
        // 11) gate: g = sigmoid(hs @ gate_w^T)
        k_gemm_wmma<2, false><<<dim3(DMODEL / 32, BT / 16), 32, 0, stream>>>(
            hsbf, gw_b + (size_t)L * DMODEL * GRUH, nullptr, gbuf, GRUH, DMODEL);
        // 12) h *= g (+ bf16 refresh for next layer)
        k_mul_gate<<<4096, 256, 0, stream>>>(h, gbuf, hbf, (size_t)BT * DMODEL);
    }

    // output head + row L2 normalization
    k_gemm_wmma<0, false><<<dim3(DIN / 32, BT / 16), 32, 0, stream>>>(
        hbf, headw_b, nullptr, out, DMODEL, DIN);
    k_normalize<<<BT / 8, 256, 0, stream>>>(out);
}
